// MLPDecoder_34419867910904
// MI455X (gfx1250) — matrix operand order repaired and checked by reading, not compiled, not run on hardware
//
#include <hip/hip_runtime.h>
#include <hip/hip_bf16.h>
#include <stdint.h>

// CDNA5 / gfx1250, wave32. WMMA bf16 16x16x32, f32 accumulate.

typedef __attribute__((ext_vector_type(8)))  __bf16 v8bf;
typedef __attribute__((ext_vector_type(16))) __bf16 v16bf;
typedef __attribute__((ext_vector_type(8)))  float  v8f;

struct U32x8 { uint4 lo, hi; };   // 32 bytes == v16bf

__device__ __forceinline__ unsigned short f32_to_bf16_rne(float f) {
  unsigned int u = __float_as_uint(f);
  u += 0x7FFFu + ((u >> 16) & 1u);          // round-to-nearest-even
  return (unsigned short)(u >> 16);
}

// 8x f32 -> 8x bf16 through the hardware convert path (v_cvt_pk_bf16_f32).
__device__ __forceinline__ v8bf cvt8_bf16(float4 a, float4 b) {
  v8f f = {a.x, a.y, a.z, a.w, b.x, b.y, b.z, b.w};
  return __builtin_convertvector(f, v8bf);
}

// ---------------------------------------------------------------------------
// Prep: pack W1 [256,128] f32 (row-major) into bf16 B-fragments in d_ws.
// Fragment order: ((n*8 + ks)*32 + lane)*16 + t, where for
// V_WMMA_F32_16X16X32_BF16 B (32x16), contiguous k order chosen here (A matches; not AMD's table):
//   N = 16*n + (lane & 15)
//   K = 32*ks + (lane >= 16 ? 16 : 0) + t,   t = 0..15 (lo half of dword = even t)
// Total: 8 n-tiles * 8 k-steps * 32 lanes * 16 bf16 = 32768 elements (64 KB).
// ---------------------------------------------------------------------------
__global__ void pack_w1_kernel(const float* __restrict__ W1,
                               unsigned short* __restrict__ wsB) {
  int tid  = blockIdx.x * blockDim.x + threadIdx.x;   // 0..32767
  int t    = tid & 15;
  int lane = (tid >> 4) & 31;
  int ks   = (tid >> 9) & 7;
  int n    = (tid >> 12) & 7;
  int col  = 16 * n + (lane & 15);
  int K    = 32 * ks + ((lane >= 16) ? 16 : 0) + t;
  wsB[tid] = f32_to_bf16_rne(W1[K * 128 + col]);
}

// ---------------------------------------------------------------------------
// Main: 8 waves/block, each wave computes 16 edges.
//  - B fragments (64 KB) staged once per block into LDS, read via ds_load_b128.
//  - A tile held entirely in VGPRs: lane owns row lane&15, K-half by lane>=16;
//    gathered directly from z (L2-resident), converted via v_cvt_pk_bf16_f32.
//  - 64x V_WMMA_F32_16X16X32_BF16 per wave; sched_group_barrier pins a
//    [3-deep ds_read prefetch][wmma] pipeline so wmma never waits dscnt 0.
//  - Register/shfl epilogue (bias + relu + 128-dot with W2), no LDS round trip.
// ---------------------------------------------------------------------------
__global__ __launch_bounds__(256) void
mlp_decoder_kernel(const float* __restrict__ z,
                   const long long* __restrict__ ei,    // [2, nE] int64
                   const float* __restrict__ b1,        // [128]
                   const float* __restrict__ W2,        // [128]
                   const float* __restrict__ b2,        // [1]
                   const uint4* __restrict__ wsB,       // packed bf16 B frags
                   float* __restrict__ out,             // [nE]
                   int nE) {
  __shared__ uint4 sB[4096];                            // 64 KB B fragments
  const int tid  = threadIdx.x;
  const int lane = tid & 31;
  const int wave = tid >> 5;
  const int e0   = blockIdx.x * 128 + wave * 16;

  // ---- stage B fragments to LDS (block-cooperative, coalesced) ----
#pragma unroll
  for (int i = 0; i < 16; ++i)
    sB[i * 256 + tid] = wsB[i * 256 + tid];

  // ---- gather this lane's A data ----
  // A layout here (16x32 bf16, contiguous k, same as B packing): lanes 0-15 & 16-31 both map to M = lane&15;
  // per k-step lane holds K = kb..kb+15, kb = 32*ks + (lane>=16)*16.
  const int row = lane & 15;
  const int h8  = (lane >> 4) * 16;
  int e = e0 + row; if (e >= nE) e = nE - 1;            // clamp; store masked later
  const long long nsrc = ei[e];
  const long long ndst = ei[nE + e];
  const float* __restrict__ srcp = z + nsrc * 128;
  const float* __restrict__ dstp = z + ndst * 128;

  v16bf afrag[8];
#pragma unroll
  for (int ks = 0; ks < 8; ++ks) {
    const float* rp = (ks < 4) ? srcp : dstp;           // concat: 0..127 src, 128..255 dst
    const int cb = 32 * (ks & 3) + h8;
    float4 f0 = *(const float4*)(rp + cb);
    float4 f1 = *(const float4*)(rp + cb + 4);
    float4 f2 = *(const float4*)(rp + cb + 8);
    float4 f3 = *(const float4*)(rp + cb + 12);
    v8bf lo = cvt8_bf16(f0, f1);                        // K = kb   .. kb+7
    v8bf hi = cvt8_bf16(f2, f3);                        // K = kb+8 .. kb+15
    afrag[ks] = __builtin_shufflevector(lo, hi,
        0, 1, 2, 3, 4, 5, 6, 7, 8, 9, 10, 11, 12, 13, 14, 15);
  }
  __syncthreads();                                      // B staged

  // ---- GEMM x(16x256) @ W1(256x128): flat 64-step loop, i = ks*8 + n ----
  // Depth-2 B prefetch through a 3-slot rotating register buffer.
  const uint4* sBl = &sB[lane * 2];
  auto loadB = [&](int i) -> U32x8 {
    const int ks = i >> 3, n = i & 7;
    const uint4* bp = sBl + ((n * 8 + ks) * 32) * 2;
    U32x8 r; r.lo = bp[0]; r.hi = bp[1];
    return r;
  };

  v8f acc[8] = {};
  U32x8 bbuf[3];
  bbuf[0] = loadB(0);
  bbuf[1] = loadB(1);
#pragma unroll
  for (int i = 0; i < 64; ++i) {
    if (i + 2 < 64) bbuf[(i + 2) % 3] = loadB(i + 2);
    v16bf b = __builtin_bit_cast(v16bf, bbuf[i % 3]);
    acc[i & 7] = __builtin_amdgcn_wmma_f32_16x16x32_bf16(
        /*neg_a=*/false, afrag[i >> 3], /*neg_b=*/false, b,
        /*c_mod=*/(short)0, acc[i & 7], /*reuse_a=*/false, /*reuse_b=*/false);
  }

#if __has_builtin(__builtin_amdgcn_sched_group_barrier)
  // Pin the pipeline shape: 3 B-fragment loads (6 ds_read) up front, then
  // repeat [1 wmma][1 fragment load (2 ds_read)] so each wmma runs with two
  // fragment loads in flight (partial s_wait_dscnt instead of 0).
  // mask 0x100 = DS read, 0x008 = MFMA/WMMA (best-effort, dep-respecting).
  __builtin_amdgcn_sched_group_barrier(0x100, 6, 0);
#pragma unroll
  for (int i = 0; i < 64; ++i) {
    __builtin_amdgcn_sched_group_barrier(0x008, 1, 0);
    __builtin_amdgcn_sched_group_barrier(0x100, 2, 0);
  }
#endif

  // ---- epilogue: out = relu(acc + b1) . W2 + b2, all in registers ----
  // C layout: acc[n] VGPR r -> M = r + 8*(lane>=16), N = lane&15, col = 16n+N.
  const int col16 = row;
  float b1v[8], w2v[8];
#pragma unroll
  for (int n = 0; n < 8; ++n) {
    b1v[n] = b1[16 * n + col16];
    w2v[n] = W2[16 * n + col16];
  }
  float p[8];
#pragma unroll
  for (int r = 0; r < 8; ++r) {
    float s = 0.0f;
#pragma unroll
    for (int n = 0; n < 8; ++n) {
      float h = acc[n][r] + b1v[n];
      h = h > 0.0f ? h : 0.0f;
      s += h * w2v[n];
    }
    p[r] = s;
  }
  // reduce across the 16-lane half (xor masks < 16 stay inside the half)
#pragma unroll
  for (int m = 1; m < 16; m <<= 1) {
#pragma unroll
    for (int r = 0; r < 8; ++r) p[r] += __shfl_xor(p[r], m);
  }
  // half0 (lane 0) holds rows 0..7, half1 (lane 16) holds rows 8..15
  if ((lane & 15) == 0) {
    const float bias = b2[0];
    const int rbase = e0 + (lane >> 4) * 8;
#pragma unroll
    for (int r = 0; r < 8; ++r) {
      int ee = rbase + r;
      if (ee < nE) out[ee] = p[r] + bias;
    }
  }
}

// ---------------------------------------------------------------------------
extern "C" void kernel_launch(void* const* d_in, const int* in_sizes, int n_in,
                              void* d_out, int out_size, void* d_ws, size_t ws_size,
                              hipStream_t stream) {
  const float*     z  = (const float*)d_in[0];
  const long long* ei = (const long long*)d_in[1];   // int64 per reference
  const float*     W1 = (const float*)d_in[2];
  const float*     b1 = (const float*)d_in[3];
  const float*     W2 = (const float*)d_in[4];
  const float*     b2 = (const float*)d_in[5];
  float* out = (float*)d_out;
  const int nE = in_sizes[1] / 2;                    // edge_label_index is [2, nE]

  // 1) pack W1 into bf16 B-fragments (64 KB of d_ws); same stream -> ordered
  pack_w1_kernel<<<128, 256, 0, stream>>>(W1, (unsigned short*)d_ws);

  // 2) decoder: 128 edges per block (8 waves x 16 edges)
  int blocks = (nE + 127) / 128;
  mlp_decoder_kernel<<<blocks, 256, 0, stream>>>(z, ei, b1, W2, b2,
                                                 (const uint4*)d_ws, out, nE);
}
